// DefaultOptimizer_66614942761508
// MI455X (gfx1250) — compile-verified, matrix operand order checked by reading, not run on hardware
//
#include <hip/hip_runtime.h>

typedef __attribute__((ext_vector_type(16))) _Float16 v16h;
typedef __attribute__((ext_vector_type(8)))  float    v8f;
typedef __attribute__((ext_vector_type(4)))  float    v4f;

#define C_DIM   4
#define M_TOTAL 65536      // B*S = 8*8192
#define K_DIM   256        // D
#define N_DIM   256        // O
#define MTILE   128
#define LDS_STRIDE 264     // 256 f16 + 8 pad

// out[c][m][n] = s_c * sum_k clamp(round(x[c][m][k]/s_c)) * w[c][n][k]
__global__ __launch_bounds__(256) void qgemm_wmma_f16(
    const float* __restrict__ x,
    const float* __restrict__ w,
    const float* __restrict__ scales,
    float* __restrict__ out)
{
    __shared__ _Float16 ldsW[N_DIM * LDS_STRIDE];   // 132 KB f16 copy of w[c]

    const int c    = blockIdx.y;
    const int m0   = blockIdx.x * MTILE;
    const int tid  = threadIdx.x;
    const int lane = tid & 31;
    const int wave = tid >> 5;
    const int wm   = wave >> 1;     // 0..3 : which 32-row M strip
    const int wn   = wave & 1;      // 0..1 : which 128-col N strip
    const int half = lane >> 4;     // 0: K{0-7,16-23}, 1: K{8-15,24-31}
    const int lr   = lane & 15;

    const float s    = scales[c];
    const float invs = 1.0f / s;

    // ---- Stage w[c] (f32 -> f16) into LDS, coalesced, 32-K-block permuted
    // {0-7,16-23,8-15,24-31} so each half-wave's B fragment is one 32B load.
    {
        const float* wc = w + (size_t)c * N_DIM * K_DIM;
        for (int idx = tid; idx < N_DIM * K_DIM / 4; idx += 256) {
            v4f v = *(const v4f*)(wc + idx * 4);
            int row = idx >> 6;
            int k   = (idx & 63) * 4;
            int sub = k & 31;
            int g   = sub >> 3;
            int gp  = ((g & 1) << 1) | (g >> 1);   // 0->0, 1->2, 2->1, 3->3
            int mapped = (k & ~31) | (gp << 3) | (sub & 7);
            _Float16* dst = &ldsW[row * LDS_STRIDE + mapped];
            dst[0] = (_Float16)v[0];
            dst[1] = (_Float16)v[1];
            dst[2] = (_Float16)v[2];
            dst[3] = (_Float16)v[3];
        }
    }
    __syncthreads();

    // ---- Per-wave: 32 M rows (2 fragments) x 128 N cols (8 tiles)
    const int rowA0 = m0 + wm * 32 + lr;
    const float* pA0 = x + ((size_t)c * M_TOTAL + rowA0) * K_DIM + half * 8;
    const float* pA1 = pA0 + 16 * (size_t)K_DIM;

    v8f acc0[8], acc1[8];
#pragma unroll
    for (int t = 0; t < 8; ++t) {
        acc0[t] = (v8f){0.f, 0.f, 0.f, 0.f, 0.f, 0.f, 0.f, 0.f};
        acc1[t] = (v8f){0.f, 0.f, 0.f, 0.f, 0.f, 0.f, 0.f, 0.f};
    }

    // double-buffered raw A (f32): offsets {0,4,16,20} within the 32-K block
    v4f r0[4], r1[4];
    {
        const int off[4] = {0, 4, 16, 20};
#pragma unroll
        for (int i = 0; i < 4; ++i) {
            r0[i] = *(const v4f*)(pA0 + off[i]);
            r1[i] = *(const v4f*)(pA1 + off[i]);
        }
    }

    // B base for this wave's N strip; tile t at +t*16 rows, k-chunk k0 at +k0
    const _Float16* ldsB = &ldsW[(wn * 128 + lr) * LDS_STRIDE + half * 16];

    v16h bcur = *(const v16h*)ldsB;      // tile 0, k0 = 0

#pragma unroll
    for (int kk = 0; kk < K_DIM / 32; ++kk) {
        const int k0 = kk * 32;

        v4f n0[4], n1[4];
        if (kk + 1 < K_DIM / 32) {
            const int off[4] = {0, 4, 16, 20};
#pragma unroll
            for (int i = 0; i < 4; ++i) {
                n0[i] = *(const v4f*)(pA0 + k0 + 32 + off[i]);
                n1[i] = *(const v4f*)(pA1 + k0 + 32 + off[i]);
            }
        }

        // quantize: exact integers in f16
        v16h aq0, aq1;
#pragma unroll
        for (int i = 0; i < 4; ++i) {
#pragma unroll
            for (int j = 0; j < 4; ++j) {
                aq0[i * 4 + j] = (_Float16)__builtin_amdgcn_fmed3f(
                    __builtin_rintf(r0[i][j] * invs), -128.f, 127.f);
                aq1[i * 4 + j] = (_Float16)__builtin_amdgcn_fmed3f(
                    __builtin_rintf(r1[i][j] * invs), -128.f, 127.f);
            }
        }

        // 8 N-tiles, rotating B buffer: next load issued before current WMMAs
#pragma unroll
        for (int t = 0; t < 8; ++t) {
            v16h bnxt = bcur;
            if (t < 7)
                bnxt = *(const v16h*)(ldsB + (t + 1) * 16 * LDS_STRIDE + k0);
            else if (kk + 1 < K_DIM / 32)
                bnxt = *(const v16h*)(ldsB + k0 + 32);
            acc0[t] = __builtin_amdgcn_wmma_f32_16x16x32_f16(
                          false, aq0, false, bcur, (short)0, acc0[t], false, false);
            acc1[t] = __builtin_amdgcn_wmma_f32_16x16x32_f16(
                          false, aq1, false, bcur, (short)0, acc1[t], false, false);
            bcur = bnxt;
        }

#pragma unroll
        for (int i = 0; i < 4; ++i) { r0[i] = n0[i]; r1[i] = n1[i]; }
    }

    // ---- Epilogue: out = s * acc
    // C/D layout: VGPR j -> lanes 0-15: (M=j, N=lane); lanes 16-31: (M=j+8, N=lane-16)
    const size_t obase = ((size_t)c * M_TOTAL + m0 + wm * 32) * N_DIM;
#pragma unroll
    for (int t = 0; t < 8; ++t) {
#pragma unroll
        for (int j = 0; j < 8; ++j) {
            int row = j + 8 * half;
            int col = wn * 128 + t * 16 + lr;
            out[obase + (size_t)row * N_DIM + col]        = s * acc0[t][j];
            out[obase + (size_t)(row + 16) * N_DIM + col] = s * acc1[t][j];
        }
    }
}

extern "C" void kernel_launch(void* const* d_in, const int* in_sizes, int n_in,
                              void* d_out, int out_size, void* d_ws, size_t ws_size,
                              hipStream_t stream) {
    (void)in_sizes; (void)n_in; (void)out_size; (void)d_ws; (void)ws_size;
    const float* x      = (const float*)d_in[0];
    const float* w      = (const float*)d_in[1];
    const float* scales = (const float*)d_in[2];
    float* out          = (float*)d_out;

    dim3 grid(M_TOTAL / MTILE, C_DIM);   // 512 x 4 workgroups
    dim3 block(256);                     // 8 wave32
    qgemm_wmma_f16<<<grid, block, 0, stream>>>(x, w, scales, out);
}
